// HiggsMultiLinear_46188078301681
// MI455X (gfx1250) — matrix operand order repaired and checked by reading, not compiled, not run on hardware
//
#include <hip/hip_runtime.h>
#include <hip/hip_bf16.h>

typedef __attribute__((ext_vector_type(16))) __bf16 v16bf;
typedef __attribute__((ext_vector_type(8)))  __bf16 v8bf;
typedef __attribute__((ext_vector_type(8)))  float  v8f;

#define K_DIM 4096
#define N_DIM 4096
#define M_DIM 4096

#define BMX 128
#define BNX 256
#define BKX 32
#define KTILES (K_DIM / BKX)
#define LDS_STRIDE 40   // 32 + 8 pad (ushort units); row pitch 80B keeps 16B alignment

__device__ __forceinline__ unsigned short f32_to_bf16(float f) {
    unsigned u = __float_as_uint(f);
    u += 0x7fffu + ((u >> 16) & 1u);           // round-to-nearest-even
    return (unsigned short)(u >> 16);
}

// ---------------------------------------------------------------------------
// Kernel 1: HIGGS dequant -> bf16 weight matrix W[4096][4096] (scales folded)
// One thread per code slot (2 weights) -> one dword store.
// ---------------------------------------------------------------------------
__global__ void higgs_dequant_kernel(const int* __restrict__ codes256,
                                     const int* __restrict__ codes16,
                                     const float* __restrict__ grid256,
                                     const float* __restrict__ grid16,
                                     const float* __restrict__ scales,
                                     unsigned int* __restrict__ Wpacked) {
    const int g = blockIdx.x * 256 + threadIdx.x;   // 0 .. 4096*2048-1
    const int n = g >> 11;                          // out-feature row
    const int j = g & 2047;                         // code column
    const float s = scales[n];
    float w0, w1;
    if (n < 2048) {
        const int c = codes256[(n << 11) + j];
        w0 = grid256[2 * c];
        w1 = grid256[2 * c + 1];
    } else {
        const int c = codes16[((n - 2048) << 11) + j];
        w0 = grid16[2 * c];
        w1 = grid16[2 * c + 1];
    }
    w0 *= s; w1 *= s;
    Wpacked[g] = (unsigned)f32_to_bf16(w0) | ((unsigned)f32_to_bf16(w1) << 16);
}

// ---------------------------------------------------------------------------
// Kernel 2: x fp32 -> bf16 (4 elements / thread)
// ---------------------------------------------------------------------------
__global__ void x_to_bf16_kernel(const float4* __restrict__ X,
                                 uint2* __restrict__ Xb) {
    const int g = blockIdx.x * 256 + threadIdx.x;
    const float4 v = X[g];
    uint2 o;
    o.x = (unsigned)f32_to_bf16(v.x) | ((unsigned)f32_to_bf16(v.y) << 16);
    o.y = (unsigned)f32_to_bf16(v.z) | ((unsigned)f32_to_bf16(v.w) << 16);
    Xb[g] = o;
}

// ---------------------------------------------------------------------------
// Kernel 3: Y = Xbf16 @ Wbf16^T + bias, fp32 accumulate via WMMA bf16.
// Block tile 128x256 (8 waves as 2x4), wave tile 64x64 (4x4 WMMA tiles),
// BK=32 (one WMMA K-step), double-buffered LDS.
// Per K-step per wave: 16 ds_load_b128 feed 16 v_wmma -> high XDL occupancy.
// ---------------------------------------------------------------------------
__global__ __launch_bounds__(256, 1)
void higgs_gemm_kernel(const unsigned short* __restrict__ Xb,
                       const unsigned short* __restrict__ Wb,
                       const float* __restrict__ bias,
                       float* __restrict__ out) {
    __shared__ unsigned short As[2][BMX][LDS_STRIDE];   // 20 KB
    __shared__ unsigned short Bs[2][BNX][LDS_STRIDE];   // 40 KB

    const int tid  = threadIdx.x;
    const int lane = tid & 31;
    const int wave = tid >> 5;
    const int wm   = wave >> 2;      // 0..1 : 64-row slab
    const int wn   = wave & 3;       // 0..3 : 64-col slab
    const int lrow = lane & 15;
    const int hi16 = lane >> 4;      // 0/1 (upper half-wave)

    const int bm = blockIdx.y * BMX;
    const int bn = blockIdx.x * BNX;

    // global->LDS copy mapping: A = 512 16B vectors, B = 1024 16B vectors,
    // 256 threads -> 2 A-vectors + 4 B-vectors per thread.
    const int ra0 = tid >> 2;            // 0..63
    const int ra1 = ra0 + 64;            // 64..127
    const int rb0 = tid >> 2;            // 0..63
    const int cv  = (tid & 3) << 3;      // element column: 0,8,16,24

    v8f acc[4][4];
    {
        v8f z = {};
        #pragma unroll
        for (int mi = 0; mi < 4; ++mi)
            #pragma unroll
            for (int ni = 0; ni < 4; ++ni)
                acc[mi][ni] = z;
    }

    uint4 ga0, ga1, gb0, gb1, gb2, gb3;

    auto load_global = [&](int k0) {
        ga0 = *(const uint4*)(Xb + (size_t)(bm + ra0) * K_DIM + k0 + cv);
        ga1 = *(const uint4*)(Xb + (size_t)(bm + ra1) * K_DIM + k0 + cv);
        gb0 = *(const uint4*)(Wb + (size_t)(bn + rb0      ) * K_DIM + k0 + cv);
        gb1 = *(const uint4*)(Wb + (size_t)(bn + rb0 +  64) * K_DIM + k0 + cv);
        gb2 = *(const uint4*)(Wb + (size_t)(bn + rb0 + 128) * K_DIM + k0 + cv);
        gb3 = *(const uint4*)(Wb + (size_t)(bn + rb0 + 192) * K_DIM + k0 + cv);
    };
    auto store_lds = [&](int buf) {
        *(uint4*)&As[buf][ra0][cv]       = ga0;
        *(uint4*)&As[buf][ra1][cv]       = ga1;
        *(uint4*)&Bs[buf][rb0][cv]       = gb0;
        *(uint4*)&Bs[buf][rb0 +  64][cv] = gb1;
        *(uint4*)&Bs[buf][rb0 + 128][cv] = gb2;
        *(uint4*)&Bs[buf][rb0 + 192][cv] = gb3;
    };

    auto compute = [&](int buf) {
        // A fragment: 16-bit A 16x32 layout — lanes 0-15 hold K {0..7,16..23},
        // lanes 16-31 hold K {8..15,24..31} for row M = lane%16.
        const int ca = hi16 << 3;        // 0 or 8
        // B fragment: 16-bit B 32x16 — lane holds column N = lane%16 with
        // lanes 0-15 K {0..7,16..23}, lanes 16-31 K {8..15,24..31}.
        const int cb = hi16 << 3;        // 0 or 8

        v16bf bfr[4];
        #pragma unroll
        for (int ni = 0; ni < 4; ++ni) {
            const unsigned short* p = &Bs[buf][wn * 64 + ni * 16 + lrow][0];
            v8bf lo = *(const v8bf*)(p + cb);
            v8bf hi = *(const v8bf*)(p + cb + 16);
            bfr[ni] = __builtin_shufflevector(lo, hi,
                      0,1,2,3,4,5,6,7,8,9,10,11,12,13,14,15);
        }
        #pragma unroll
        for (int mi = 0; mi < 4; ++mi) {
            const unsigned short* p = &As[buf][wm * 64 + mi * 16 + lrow][0];
            v8bf lo = *(const v8bf*)(p + ca);
            v8bf hi = *(const v8bf*)(p + ca + 16);
            v16bf af = __builtin_shufflevector(lo, hi,
                       0,1,2,3,4,5,6,7,8,9,10,11,12,13,14,15);
            #pragma unroll
            for (int ni = 0; ni < 4; ++ni)
                acc[mi][ni] = __builtin_amdgcn_wmma_f32_16x16x32_bf16(
                    /*neg_a=*/false, af,
                    /*neg_b=*/false, bfr[ni],
                    /*c_mod=*/(short)0, acc[mi][ni],
                    /*reuse_a=*/false, /*reuse_b=*/false);
        }
    };

    // prologue: stage tile 0
    load_global(0);
    store_lds(0);
    __syncthreads();

    for (int ks = 0; ks < KTILES; ++ks) {
        const int cur = ks & 1;
        const int nxt = cur ^ 1;
        const bool more = (ks + 1) < KTILES;
        if (more) load_global((ks + 1) * BKX);
        if (ks + 2 < KTILES) {
            __builtin_prefetch(Xb + (size_t)(bm + ra0) * K_DIM + (ks + 2) * BKX + cv, 0, 0);
            __builtin_prefetch(Wb + (size_t)(bn + rb0) * K_DIM + (ks + 2) * BKX + cv, 0, 0);
        }
        compute(cur);
        if (more) store_lds(nxt);
        __syncthreads();
    }

    // Epilogue: C/D 16x16 f32 layout -> element (r + 8*hi16, lane%16) in VGPR r.
    const int mbase = bm + wm * 64 + hi16 * 8;
    #pragma unroll
    for (int mi = 0; mi < 4; ++mi) {
        #pragma unroll
        for (int ni = 0; ni < 4; ++ni) {
            const int col = bn + wn * 64 + ni * 16 + lrow;
            const float bv = bias[col];
            #pragma unroll
            for (int r = 0; r < 8; ++r) {
                const int row = mbase + mi * 16 + r;
                out[(size_t)row * N_DIM + col] = acc[mi][ni][r] + bv;
            }
        }
    }
}

extern "C" void kernel_launch(void* const* d_in, const int* in_sizes, int n_in,
                              void* d_out, int out_size, void* d_ws, size_t ws_size,
                              hipStream_t stream) {
    const float* x        = (const float*)d_in[0];
    const int*   c256     = (const int*)d_in[1];
    const int*   c16      = (const int*)d_in[2];
    const float* g256     = (const float*)d_in[3];
    const float* g16      = (const float*)d_in[4];
    const float* scales   = (const float*)d_in[5];
    const float* bias     = (const float*)d_in[6];
    float*       out      = (float*)d_out;

    // workspace: W_bf16 (32 MB) then X_bf16 (32 MB)
    unsigned short* Wb = (unsigned short*)d_ws;
    unsigned short* Xb = (unsigned short*)((char*)d_ws + (size_t)N_DIM * K_DIM * 2);

    // 1) dequant codes -> bf16 weights (scales folded in)
    higgs_dequant_kernel<<<(N_DIM * (K_DIM / 2)) / 256, 256, 0, stream>>>(
        c256, c16, g256, g16, scales, (unsigned int*)Wb);

    // 2) x fp32 -> bf16
    x_to_bf16_kernel<<<((size_t)M_DIM * K_DIM / 4) / 256, 256, 0, stream>>>(
        (const float4*)x, (uint2*)Xb);

    // 3) WMMA GEMM + bias
    dim3 grid(N_DIM / BNX, M_DIM / BMX);
    higgs_gemm_kernel<<<grid, 256, 0, stream>>>(Xb, Wb, bias, out);
}
